// AnyUp_21938692947992
// MI455X (gfx1250) — compile-verified, matrix operand order checked by reading, not run on hardware
//
#include <hip/hip_runtime.h>
#include <hip/hip_bf16.h>
#include <math.h>

// ---------------------------------------------------------------------------
// AnyUp forward for MI455X (gfx1250, wave32, WMMA).
//
// Input ordering assumption (setup_inputs dict insertion order, params nested
// in insertion order):
//   d_in[0] = image    (2,3,1,128,128) f32
//   d_in[1] = features (2,384,1,32,32) f32
//   d_in[2..] = params for encoders in order: img, query, key, lfu, agg.
//   Each encoder block (18 tensors):
//     +0 pre_w, +1 pre_b,
//     rb0: +2 gn1_s, +3 gn1_b, +4 w1, +5 b1, +6 gn2_s, +7 gn2_b, +8 w2, +9 b2
//     rb1: +10..+17 (same order)
// Output: (2,384,1,128,128) f32.
// ---------------------------------------------------------------------------

typedef __attribute__((ext_vector_type(16))) __bf16 v16bf;
typedef __attribute__((ext_vector_type(8)))  __bf16 v8bf;
typedef __attribute__((ext_vector_type(8)))  float  v8f;

#define BATCH 2
#define HW_F  16384   // 128*128
#define HW_C  1024    // 32*32

__device__ __forceinline__ int reflect_idx(int i, int n) {
  if (i < 0) i = -i;
  if (i >= n) i = 2 * n - 2 - i;
  return i;
}
__device__ __forceinline__ float silu_f(float x) { return x / (1.f + __expf(-x)); }

// Load a 16(row)x32(K) bf16 fragment per CDNA5 16-bit A/B layout:
// lane&15 selects the row; per-VGPR K pairs: v0..3 -> K {0..7}|{8..15} by lane
// half, v4..7 -> +16. rowk0 must point at (row, K=0), contiguous in K.
__device__ __forceinline__ v16bf load_frag16x32(const __bf16* rowk0, int lane) {
  v16bf f;
  const int khalf = (lane & 16) ? 8 : 0;
#pragma unroll
  for (int v = 0; v < 8; ++v) {
    const int k = ((v & 3) << 1) + khalf + ((v & 4) ? 16 : 0);
    f[2 * v]     = rowk0[k];
    f[2 * v + 1] = rowk0[k + 1];
  }
  return f;
}

// ------------------------------ pack kernels -------------------------------

// w: [128, cin, taps] f32  ->  wp: [128, taps*cinP] bf16, k-order (tap, ci)
__global__ void pack_weights(const float* __restrict__ w, __bf16* __restrict__ wp,
                             int cin, int cinP, int taps) {
  int idx = blockIdx.x * blockDim.x + threadIdx.x;
  int K = taps * cinP;
  if (idx >= 128 * K) return;
  int co = idx / K, k = idx - co * K;
  int tap = k / cinP, ci = k - tap * cinP;
  float v = (ci < cin) ? w[((size_t)co * cin + ci) * taps + tap] : 0.f;
  wp[idx] = (__bf16)v;
}

// image (2,3,16384) f32 -> (2,16384,32) bf16 (zero padded channels)
__global__ void pack_image(const float* __restrict__ img, __bf16* __restrict__ xp) {
  int idx = blockIdx.x * blockDim.x + threadIdx.x;
  if (idx >= BATCH * HW_F * 32) return;
  int b = idx / (HW_F * 32), r = idx - b * (HW_F * 32);
  int pos = r >> 5, c = r & 31;
  float v = (c < 3) ? img[((size_t)b * 3 + c) * HW_F + pos] : 0.f;
  xp[idx] = (__bf16)v;
}

// f32 channel-major [B,C,HW] -> bf16 token-major [B,HW,C]
__global__ void pack_tokmajor(const float* __restrict__ x, __bf16* __restrict__ xp,
                              int C, int HW) {
  int idx = blockIdx.x * blockDim.x + threadIdx.x;
  if (idx >= BATCH * C * HW) return;
  int b = idx / (C * HW), r = idx - b * (C * HW);
  int pos = r / C, c = r - pos * C;
  xp[idx] = (__bf16)x[((size_t)b * C + c) * HW + pos];
}

// plain f32 -> bf16 (same layout)
__global__ void pack_bf16(const float* __restrict__ x, __bf16* __restrict__ xp, int n) {
  int idx = blockIdx.x * blockDim.x + threadIdx.x;
  if (idx < n) xp[idx] = (__bf16)x[idx];
}

// concat [kpool(128ch), kf(128ch)] channel-major f32 -> bf16 token-major [B,1024,256]
__global__ void concat_pack(const float* __restrict__ k1, const float* __restrict__ k2,
                            __bf16* __restrict__ xp) {
  int idx = blockIdx.x * blockDim.x + threadIdx.x;
  if (idx >= BATCH * HW_C * 256) return;
  int b = idx / (HW_C * 256), r = idx - b * (HW_C * 256);
  int pos = r >> 8, c = r & 255;
  float v = (c < 128) ? k1[((size_t)b * 128 + c) * HW_C + pos]
                      : k2[((size_t)b * 128 + (c - 128)) * HW_C + pos];
  xp[idx] = (__bf16)v;
}

// features (2,384,1024) f32 -> L2-normalized over C, bf16 token-major (2,1024,384)
__global__ void featnorm_pack(const float* __restrict__ f, __bf16* __restrict__ fp) {
  int idx = blockIdx.x * blockDim.x + threadIdx.x;
  if (idx >= BATCH * HW_C) return;
  int b = idx >> 10, pos = idx & 1023;
  const float* src = f + (size_t)b * 384 * HW_C + pos;
  float s = 0.f;
  for (int c = 0; c < 384; ++c) { float v = src[(size_t)c * HW_C]; s += v * v; }
  float inv = 1.f / fmaxf(sqrtf(s), 1e-12f);
  __bf16* o = fp + (size_t)idx * 384;
  for (int c = 0; c < 384; ++c) o[c] = (__bf16)(src[(size_t)c * HW_C] * inv);
}

// ------------------------------ group norm ---------------------------------

// One block per (b, group). x: [B,128,HW]; stats[(b*8+g)*2] = {mu, rstd}
__global__ void gn_stats(const float* __restrict__ x, float* __restrict__ stats, int HW) {
  int bg = blockIdx.x, b = bg >> 3, g = bg & 7;
  const float* base = x + ((size_t)b * 128 + g * 16) * HW;
  int n = 16 * HW;
  float s = 0.f, s2 = 0.f;
  for (int i = threadIdx.x; i < n; i += blockDim.x) {
    float v = base[i]; s += v; s2 += v * v;
  }
  __shared__ float shs[256], shq[256];
  shs[threadIdx.x] = s; shq[threadIdx.x] = s2;
  __syncthreads();
  for (int o = 128; o > 0; o >>= 1) {
    if (threadIdx.x < (unsigned)o) {
      shs[threadIdx.x] += shs[threadIdx.x + o];
      shq[threadIdx.x] += shq[threadIdx.x + o];
    }
    __syncthreads();
  }
  if (threadIdx.x == 0) {
    float mu = shs[0] / n;
    float var = shq[0] / n - mu * mu;
    stats[bg * 2]     = mu;
    stats[bg * 2 + 1] = rsqrtf(var + 1e-5f);
  }
}

// GN + SiLU + pack: f32 cm [B,128,HW] -> bf16 token-major [B,HW,128]
__global__ void gn_silu_pack(const float* __restrict__ x, const float* __restrict__ stats,
                             const float* __restrict__ sc, const float* __restrict__ bi,
                             __bf16* __restrict__ xp, int HW) {
  int idx = blockIdx.x * blockDim.x + threadIdx.x;
  if (idx >= BATCH * 128 * HW) return;
  int b = idx / (128 * HW), r = idx - b * (128 * HW);
  int pos = r >> 7, c = r & 127, g = c >> 4;
  float mu = stats[(b * 8 + g) * 2], rs = stats[(b * 8 + g) * 2 + 1];
  float v = (x[((size_t)b * 128 + c) * HW + pos] - mu) * rs * sc[c] + bi[c];
  xp[idx] = (__bf16)silu_f(v);
}

// ------------------------- implicit-GEMM conv (WMMA) -----------------------
// out[b,co,pos] = sum_{tap,ci} Wp[co, tap*cinP+ci] * Xp[b, reflect(pos,tap), ci] + bias
// Block: 256 threads = 8 waves. Block tile = 128(M, all couts) x 64(N, positions).
// Per K-chunk of 32: activation tile (64 pos x 32 k) is staged once in LDS
// (16B per-thread loads, row stride padded to 40 elems for bank spread), then
// each wave runs 1 A-fragment x 4 B-fragments = 4 WMMAs. Weights stream from
// L2 (contiguous K pairs). Reflect padding resolved at staging time.
__global__ void conv_wmma(const __bf16* __restrict__ xp, const __bf16* __restrict__ wp,
                          const float* __restrict__ bias, const float* __restrict__ skip,
                          float* __restrict__ out,
                          int H, int W, int cinP, int kh, int pad) {
  const int HWp = H * W;
  const int b = blockIdx.y;
  const int lane = threadIdx.x & 31;
  const int wave = threadIdx.x >> 5;
  const int mbase = wave * 16;
  const int n = lane & 15;
  const int nbase = blockIdx.x * 64;
  const int K = kh * kh * cinP;

  // staging role: thread -> (local position, 8-elem k offset)
  const int spos  = threadIdx.x & 63;
  const int skoff = (threadIdx.x >> 6) << 3;   // 0,8,16,24
  const int gpos  = nbase + spos;
  const int gy = gpos / W, gx = gpos - gy * W;

  __shared__ __bf16 tileB[64][40];             // 40: pad for LDS bank spread

  const __bf16* wrow = wp + (size_t)(mbase + n) * K;  // A row (cout) = lane&15
  const v8f vz = {0.f, 0.f, 0.f, 0.f, 0.f, 0.f, 0.f, 0.f};
  v8f acc[4];
#pragma unroll
  for (int j = 0; j < 4; ++j) acc[j] = vz;

  for (int kc = 0; kc < K; kc += 32) {
    int tap = kc / cinP;                       // cinP is a multiple of 32
    int ciB = kc - tap * cinP + skoff;
    int th  = tap / kh;
    int dy = th - pad, dx = tap - th * kh - pad;
    int sy = reflect_idx(gy + dy, H), sx = reflect_idx(gx + dx, W);
    const __bf16* src = xp + ((size_t)b * HWp + sy * W + sx) * cinP + ciB;
    v8bf ld = *(const v8bf*)src;               // 16B global load
    __builtin_prefetch(src + 32, 0, 0);        // next K-chunk -> global_prefetch_b8
    __syncthreads();                           // WAR vs previous chunk's reads
    *(v8bf*)&tileB[spos][skoff] = ld;
    __syncthreads();
    v16bf af = load_frag16x32(wrow + kc, lane);
#pragma unroll
    for (int j = 0; j < 4; ++j) {
      v16bf bf = load_frag16x32(&tileB[j * 16 + n][0], lane);
      acc[j] = __builtin_amdgcn_wmma_f32_16x16x32_bf16(false, af, false, bf,
                                                       (short)0, acc[j], false, false);
    }
  }

  const int hi = (lane >> 4) & 1;
#pragma unroll
  for (int j = 0; j < 4; ++j) {
    int pos = nbase + j * 16 + n;
#pragma unroll
    for (int r = 0; r < 8; ++r) {
      int m = mbase + r + hi * 8;
      size_t o = ((size_t)b * 128 + m) * HWp + pos;
      float v = acc[j][r] + bias[m];
      if (skip) v += skip[o];
      out[o] = v;
    }
  }
}

// ------------------------------ RoPE + pack --------------------------------
// enc f32 cm [2,128,16384] -> bf16 token-major [2,16384,128] with 3D RoPE.
__global__ void rope_pack(const float* __restrict__ enc, __bf16* __restrict__ rp) {
  int idx = blockIdx.x * blockDim.x + threadIdx.x;
  if (idx >= BATCH * HW_F) return;
  int b = idx >> 14, pos = idx & 16383;
  float crd[3] = {0.f, (float)(pos >> 7), (float)(pos & 127)};
  const float* xin = enc + (size_t)b * 128 * HW_F + pos;  // channel stride HW_F
  __bf16* o = rp + (size_t)idx * 128;
  const float nlog = -9.210340371976184f / 21.f;          // -ln(10000)/half
#pragma unroll 1
  for (int ax = 0; ax < 3; ++ax) {
    for (int j = 0; j < 21; ++j) {
      float ang = crd[ax] * __expf(nlog * (float)j);
      float s, c;
      __sincosf(ang, &s, &c);
      float x1 = xin[(size_t)(ax * 42 + j) * HW_F];
      float x2 = xin[(size_t)(ax * 42 + 21 + j) * HW_F];
      o[ax * 42 + j]      = (__bf16)(x1 * c - x2 * s);
      o[ax * 42 + 21 + j] = (__bf16)(x1 * s + x2 * c);
    }
  }
  o[126] = (__bf16)xin[(size_t)126 * HW_F];
  o[127] = (__bf16)xin[(size_t)127 * HW_F];
}

// ------------------------------ 4x4 avg pool -------------------------------
__global__ void pool4(const float* __restrict__ in, float* __restrict__ out) {
  int idx = blockIdx.x * blockDim.x + threadIdx.x;
  if (idx >= BATCH * 128 * HW_C) return;
  int bc = idx >> 10, p = idx & 1023;
  int ky = p >> 5, kx = p & 31;
  const float* base = in + (size_t)bc * HW_F + (ky * 4) * 128 + kx * 4;
  float s = 0.f;
#pragma unroll
  for (int i = 0; i < 4; ++i)
#pragma unroll
    for (int j = 0; j < 4; ++j) s += base[i * 128 + j];
  out[idx] = s * (1.f / 16.f);
}

// --------------------- windowed flash attention (WMMA) ---------------------
// grid (128 y-rows, 4 heads, 2 batch); 8 waves/block, each wave = 16 queries
// (one x-span of a fixed y). Online softmax; whole 32-key rows (one ky) are
// skipped when outside the 0.1 window (uniform per block: all waves share y).
__global__ void attn_wmma(const __bf16* __restrict__ Qp,  // [2,16384,128] tok-major
                          const __bf16* __restrict__ Kp,  // [2,1024,128]  tok-major
                          const __bf16* __restrict__ Vp,  // [2,384,1024]  ch-major
                          float* __restrict__ out) {      // [2,384,16384]
  const int b = blockIdx.z, h = blockIdx.y, yq = blockIdx.x;
  const int lane = threadIdx.x & 31, wave = threadIdx.x >> 5;
  const int xbase = wave * 16;
  const int qbase = yq * 128 + xbase;
  const int n = lane & 15;
  const int hi = (lane >> 4) & 1;
  const float cy = (yq + 0.5f) * (1.f / 128.f);
  const float win = 0.1f + 1e-6f;
  const float scale = 0.17677669529663687f;  // 1/sqrt(32)

  const __bf16* qrow = Qp + ((size_t)(b * HW_F + qbase + n) * 128) + h * 32;
  v16bf qf = load_frag16x32(qrow, lane);

  const v8f vz = {0.f, 0.f, 0.f, 0.f, 0.f, 0.f, 0.f, 0.f};
  v8f acc[6];
#pragma unroll
  for (int j = 0; j < 6; ++j) acc[j] = vz;
  float rowM[8], rowL[8];
#pragma unroll
  for (int r = 0; r < 8; ++r) { rowM[r] = -INFINITY; rowL[r] = 0.f; }

  __shared__ __bf16 pShared[8][16][32];
  __bf16 (*pS)[32] = pShared[wave];

  for (int ky = 0; ky < 32; ++ky) {
    float cky = (ky + 0.5f) * (1.f / 32.f);
    if (fabsf(cy - cky) > win) continue;  // uniform across block (same y)
    const int kb = ky * 32;

    const __bf16* kr0 = Kp + ((size_t)(b * HW_C + kb + n) * 128) + h * 32;
    const __bf16* kr1 = Kp + ((size_t)(b * HW_C + kb + 16 + n) * 128) + h * 32;
    __builtin_prefetch(Vp + ((size_t)(b * 384 + h * 96 + n) * HW_C) + kb, 0, 0);
    v16bf kf0 = load_frag16x32(kr0, lane);
    v16bf kf1 = load_frag16x32(kr1, lane);
    v8f s0 = __builtin_amdgcn_wmma_f32_16x16x32_bf16(false, qf, false, kf0,
                                                     (short)0, vz, false, false);
    v8f s1 = __builtin_amdgcn_wmma_f32_16x16x32_bf16(false, qf, false, kf1,
                                                     (short)0, vz, false, false);

    const float ckx0 = (n + 0.5f) * (1.f / 32.f);
    const float ckx1 = (n + 16 + 0.5f) * (1.f / 32.f);
#pragma unroll
    for (int r = 0; r < 8; ++r) {
      int m = r + hi * 8;
      float cxq = (xbase + m + 0.5f) * (1.f / 128.f);
      float v0 = s0[r] * scale, v1 = s1[r] * scale;
      if (fabsf(cxq - ckx0) > win) v0 = -1e30f;
      if (fabsf(cxq - ckx1) > win) v1 = -1e30f;
      float cm = fmaxf(v0, v1);
#pragma unroll
      for (int off = 1; off < 16; off <<= 1)
        cm = fmaxf(cm, __shfl_xor(cm, off, 32));      // half-wave row reduce
      float nM = fmaxf(rowM[r], cm);
      float alpha = __expf(rowM[r] - nM);
      float p0 = __expf(v0 - nM), p1 = __expf(v1 - nM);
      float ps = p0 + p1;
#pragma unroll
      for (int off = 1; off < 16; off <<= 1) ps += __shfl_xor(ps, off, 32);
      rowL[r] = rowL[r] * alpha + ps;
      rowM[r] = nM;
#pragma unroll
      for (int j = 0; j < 6; ++j) acc[j][r] *= alpha;
      pS[m][n]      = (__bf16)p0;
      pS[m][n + 16] = (__bf16)p1;
    }
    asm volatile("s_wait_dscnt 0x0" ::: "memory");  // LDS write->read, same wave

    v16bf pf = load_frag16x32(&pS[n][0], lane);     // P as A-fragment (16x32)
#pragma unroll
    for (int j = 0; j < 6; ++j) {
      const __bf16* vrow = Vp + ((size_t)(b * 384 + h * 96 + j * 16 + n) * HW_C) + kb;
      v16bf vf = load_frag16x32(vrow, lane);
      acc[j] = __builtin_amdgcn_wmma_f32_16x16x32_bf16(false, pf, false, vf,
                                                       (short)0, acc[j], false, false);
    }
  }

#pragma unroll
  for (int r = 0; r < 8; ++r) {
    int m = r + hi * 8;
    float inv = 1.f / rowL[r];
    size_t q = (size_t)yq * 128 + xbase + m;
#pragma unroll
    for (int j = 0; j < 6; ++j) {
      int c = h * 96 + j * 16 + n;
      out[((size_t)(b * 384 + c)) * HW_F + q] = acc[j][r] * inv;
    }
  }
}

// ------------------------------ host orchestration -------------------------

namespace {

struct Ctx {
  hipStream_t s;
  void* const* din;
  __bf16* wp;
  float* stats;
};

inline const float* F(const Ctx& c, int i) { return (const float*)c.din[i]; }

inline void run_conv(const Ctx& c, const __bf16* xp, const float* w, const float* bias,
                     const float* skip, float* out,
                     int H, int W, int cin, int cinP, int kh, int pad) {
  int taps = kh * kh;
  int tot = 128 * taps * cinP;
  pack_weights<<<(tot + 255) / 256, 256, 0, c.s>>>(w, c.wp, cin, cinP, taps);
  dim3 g((H * W) / 64, BATCH);
  conv_wmma<<<g, 256, 0, c.s>>>(xp, c.wp, bias, skip, out, H, W, cinP, kh, pad);
}

// out = conv2(silu(gn2(conv1(silu(gn1(x)))))) + x ; x preserved, out != x.
inline void run_resblock(const Ctx& c, float* x, float* out, __bf16* pbuf,
                         int H, int W, int pb) {
  int HW = H * W;
  int tot = BATCH * 128 * HW;
  gn_stats<<<BATCH * 8, 256, 0, c.s>>>(x, c.stats, HW);
  gn_silu_pack<<<(tot + 255) / 256, 256, 0, c.s>>>(x, c.stats, F(c, pb + 0),
                                                   F(c, pb + 1), pbuf, HW);
  run_conv(c, pbuf, F(c, pb + 2), F(c, pb + 3), nullptr, out, H, W, 128, 128, 1, 0);
  gn_stats<<<BATCH * 8, 256, 0, c.s>>>(out, c.stats, HW);
  gn_silu_pack<<<(tot + 255) / 256, 256, 0, c.s>>>(out, c.stats, F(c, pb + 4),
                                                   F(c, pb + 5), pbuf, HW);
  run_conv(c, pbuf, F(c, pb + 6), F(c, pb + 7), x, out, H, W, 128, 128, 1, 0);
}

inline char* bump(char*& p, size_t bytes) {
  char* r = p;
  p += (bytes + 255) & ~(size_t)255;
  return r;
}

}  // namespace

extern "C" void kernel_launch(void* const* d_in, const int* in_sizes, int n_in,
                              void* d_out, int out_size, void* d_ws, size_t ws_size,
                              hipStream_t stream) {
  (void)in_sizes; (void)n_in; (void)out_size; (void)ws_size;
  const float* image    = (const float*)d_in[0];
  const float* features = (const float*)d_in[1];
  const int ENC_IMG = 2, ENC_Q = 20, ENC_K = 38, ENC_LFU = 56, ENC_AGG = 74;

  char* p = (char*)d_ws;
  float*  E0    = (float*) bump(p, (size_t)BATCH * 128 * HW_F * 4);
  float*  E1    = (float*) bump(p, (size_t)BATCH * 128 * HW_F * 4);
  __bf16* P     = (__bf16*)bump(p, (size_t)BATCH * HW_F * 128 * 2);
  __bf16* Rp    = (__bf16*)bump(p, (size_t)BATCH * HW_F * 128 * 2);
  __bf16* Qp    = (__bf16*)bump(p, (size_t)BATCH * HW_F * 128 * 2);
  __bf16* Ip    = (__bf16*)bump(p, (size_t)BATCH * HW_F * 32 * 2);
  float*  S0    = (float*) bump(p, (size_t)BATCH * 128 * HW_C * 4);
  float*  S1    = (float*) bump(p, (size_t)BATCH * 128 * HW_C * 4);
  float*  KPOOL = (float*) bump(p, (size_t)BATCH * 128 * HW_C * 4);
  __bf16* Fp    = (__bf16*)bump(p, (size_t)BATCH * HW_C * 384 * 2);
  __bf16* Ap    = (__bf16*)bump(p, (size_t)BATCH * HW_C * 256 * 2);
  __bf16* SPp   = (__bf16*)bump(p, (size_t)BATCH * HW_C * 128 * 2);
  __bf16* Kp    = (__bf16*)bump(p, (size_t)BATCH * HW_C * 128 * 2);
  __bf16* Vp    = (__bf16*)bump(p, (size_t)BATCH * 384 * HW_C * 2);
  __bf16* Wp    = (__bf16*)bump(p, (size_t)128 * 9600 * 2);
  float*  stats = (float*) bump(p, 64 * 4);

  Ctx c{stream, d_in, Wp, stats};

  // 1) image encoder @128x128
  {
    int tot = BATCH * HW_F * 32;
    pack_image<<<(tot + 255) / 256, 256, 0, stream>>>(image, Ip);
  }
  run_conv(c, Ip, F(c, ENC_IMG + 0), F(c, ENC_IMG + 1), nullptr, E0,
           128, 128, 3, 32, 1, 0);
  run_resblock(c, E0, E1, P, 128, 128, ENC_IMG + 2);
  run_resblock(c, E1, E0, P, 128, 128, ENC_IMG + 10);

  // 2) RoPE + pack -> Rp
  rope_pack<<<(BATCH * HW_F + 255) / 256, 256, 0, stream>>>(E0, Rp);

  // 3) query encoder -> Qp
  run_conv(c, Rp, F(c, ENC_Q + 0), F(c, ENC_Q + 1), nullptr, E1,
           128, 128, 128, 128, 1, 0);
  run_resblock(c, E1, E0, P, 128, 128, ENC_Q + 2);
  run_resblock(c, E0, E1, P, 128, 128, ENC_Q + 10);
  {
    int tot = BATCH * 128 * HW_F;
    pack_tokmajor<<<(tot + 255) / 256, 256, 0, stream>>>(E1, Qp, 128, HW_F);
  }

  // 4) key encoder -> pool -> KPOOL
  run_conv(c, Rp, F(c, ENC_K + 0), F(c, ENC_K + 1), nullptr, E0,
           128, 128, 128, 128, 1, 0);
  run_resblock(c, E0, E1, P, 128, 128, ENC_K + 2);
  run_resblock(c, E1, E0, P, 128, 128, ENC_K + 10);
  {
    int tot = BATCH * 128 * HW_C;
    pool4<<<(tot + 255) / 256, 256, 0, stream>>>(E0, KPOOL);
  }

  // 5) lfu encoder @32x32 (5x5 reflect pad 2 on L2-normalized features)
  featnorm_pack<<<(BATCH * HW_C + 255) / 256, 256, 0, stream>>>(features, Fp);
  run_conv(c, Fp, F(c, ENC_LFU + 0), F(c, ENC_LFU + 1), nullptr, S0,
           32, 32, 384, 384, 5, 2);
  run_resblock(c, S0, S1, SPp, 32, 32, ENC_LFU + 2);
  run_resblock(c, S1, S0, SPp, 32, 32, ENC_LFU + 10);  // kf = S0

  // 6) agg encoder (concat -> 3x3 reflect pad 1)
  {
    int tot = BATCH * HW_C * 256;
    concat_pack<<<(tot + 255) / 256, 256, 0, stream>>>(KPOOL, S0, Ap);
  }
  run_conv(c, Ap, F(c, ENC_AGG + 0), F(c, ENC_AGG + 1), nullptr, S1,
           32, 32, 256, 256, 3, 1);
  run_resblock(c, S1, S0, SPp, 32, 32, ENC_AGG + 2);
  run_resblock(c, S0, S1, SPp, 32, 32, ENC_AGG + 10);  // k = S1
  {
    int tot = BATCH * 128 * HW_C;
    pack_tokmajor<<<(tot + 255) / 256, 256, 0, stream>>>(S1, Kp, 128, HW_C);
  }

  // 7) V pack (bf16, channel-major)
  {
    int tot = BATCH * 384 * HW_C;
    pack_bf16<<<(tot + 255) / 256, 256, 0, stream>>>(features, Vp, tot);
  }

  // 8) windowed flash attention
  {
    dim3 g(128, 4, BATCH);
    attn_wmma<<<g, 256, 0, stream>>>(Qp, Kp, Vp, (float*)d_out);
  }
}
